// AttentionHyperedgeSelector_17549236371598
// MI455X (gfx1250) — compile-verified, matrix operand order checked by reading, not run on hardware
//
#include <hip/hip_runtime.h>
#include <hip/hip_bf16.h>
#include <math.h>

typedef __attribute__((ext_vector_type(16))) _Float16 v16h;
typedef __attribute__((ext_vector_type(8)))  float    v8f;

#define EPB   16   // edges per block
#define D_IMG 64
#define D_TXT 128
#define HID   64

__device__ __forceinline__ int lower_bound_i32(const int* __restrict__ arr, int n, int val) {
    int lo = 0, hi = n;
    while (lo < hi) {
        int mid = (lo + hi) >> 1;
        if (arr[mid] < val) lo = mid + 1; else hi = mid;
    }
    return lo;
}

// K index of element i (0..15) of the v16h fragment for a lane in group grp
// (grp = lane>>4), per CDNA5 ISA 16-bit A/B 16x32 layout:
//   VGPR v (=i>>1), halves hold K = (v>=4?16:0) + 8*grp + 2*(v&3) + (i&1)
__device__ __forceinline__ int wmma_kidx(int i, int grp) {
    int v = i >> 1;
    return ((v >= 4) ? 16 : 0) + 8 * grp + 2 * (v & 3) + (i & 1);
}

__global__ __launch_bounds__(256)
void AttentionHyperedgeSelector_kernel(
    const float* __restrict__ feat_img, const float* __restrict__ feat_txt,
    const int* __restrict__ node_idx,   const int* __restrict__ seg_id,
    const float* __restrict__ W1i, const float* __restrict__ b1i,
    const float* __restrict__ W2i, const float* __restrict__ b2i,
    const float* __restrict__ W1t, const float* __restrict__ b1t,
    const float* __restrict__ W2t, const float* __restrict__ b2t,
    const float* __restrict__ alpha,
    float* __restrict__ out, int n_edges, int n_memb)
{
    __shared__ float pooled_img[EPB][D_IMG];   // 4 KB
    __shared__ float pooled_txt[EPB][D_TXT];   // 8 KB
    __shared__ int   seg_lo[EPB + 1];
    __shared__ float inv_cnt[EPB];
    __shared__ float s_modal[2][EPB];

    const int tid = threadIdx.x;
    const int e0  = blockIdx.x * EPB;

    // ---- 1) segment boundaries via binary search on sorted seg_id ----
    if (tid <= EPB) {
        seg_lo[tid] = lower_bound_i32(seg_id, n_memb, e0 + tid);
    }
    // ---- 2) zero pooled accumulators ----
    for (int i = tid; i < EPB * D_IMG; i += 256) ((float*)pooled_img)[i] = 0.f;
    for (int i = tid; i < EPB * D_TXT; i += 256) ((float*)pooled_txt)[i] = 0.f;
    __syncthreads();

    // ---- 3) gather + segment-sum into LDS (all 8 waves) ----
    {   // image: thread = (member-row mr in 0..3) x (feature f in 0..63)
        const int f  = tid & (D_IMG - 1);
        const int mr = tid >> 6;
        for (int le = 0; le < EPB; ++le) {
            const int lo = seg_lo[le], hi = seg_lo[le + 1];
            float s = 0.f;
            for (int j = lo + mr; j < hi; j += 4) {
                const int node = node_idx[j];
                s += feat_img[(long long)node * D_IMG + f];
            }
            if (s != 0.f) atomicAdd(&pooled_img[le][f], s);
        }
    }
    {   // text: thread = (mr in 0..1) x (f in 0..127)
        const int f  = tid & (D_TXT - 1);
        const int mr = tid >> 7;
        for (int le = 0; le < EPB; ++le) {
            const int lo = seg_lo[le], hi = seg_lo[le + 1];
            float s = 0.f;
            for (int j = lo + mr; j < hi; j += 2) {
                const int node = node_idx[j];
                s += feat_txt[(long long)node * D_TXT + f];
            }
            if (s != 0.f) atomicAdd(&pooled_txt[le][f], s);
        }
    }
    if (tid < EPB) {
        const int c = seg_lo[tid + 1] - seg_lo[tid];
        inv_cnt[tid] = 1.0f / (float)(c > 1 ? c : 1);
    }
    __syncthreads();

    // ---- 4) WMMA MLPs on wave 0 (EXEC all-ones within the wave) ----
    if (tid < 32) {
        const int lane = tid;
        const int mc   = lane & 15;   // column for B/C-D fragments, row for A
        const int grp  = lane >> 4;

        // ===== image MLP: [16,64] @ [64,64] -> relu -> @ [64] =====
        {
            float pr[8];
            #pragma unroll
            for (int r = 0; r < 8; ++r) pr[r] = 0.f;
            #pragma unroll
            for (int t = 0; t < 4; ++t) {           // N tiles of 16
                const int n0 = t * 16;
                v8f c = {};
                #pragma unroll
                for (int kb = 0; kb < 2; ++kb) {    // K steps of 32
                    v16h a, b;
                    #pragma unroll
                    for (int i = 0; i < 16; ++i) {
                        const int k = kb * 32 + wmma_kidx(i, grp);
                        a[i] = (_Float16)(pooled_img[mc][k] * inv_cnt[mc]);
                        b[i] = (_Float16)(W1i[k * HID + n0 + mc]);
                    }
                    c = __builtin_amdgcn_wmma_f32_16x16x32_f16(
                            false, a, false, b, (short)0, c, false, false);
                }
                const float w2v = W2i[n0 + mc];
                const float b1v = b1i[n0 + mc];
                #pragma unroll
                for (int r = 0; r < 8; ++r) {
                    float h = c[r] + b1v;
                    h = h > 0.f ? h : 0.f;          // relu
                    pr[r] += h * w2v;               // layer-2 dot, partial over this lane's column
                }
            }
            // reduce over 16 columns held by lanes within each half (C rows M=r+8*grp)
            #pragma unroll
            for (int r = 0; r < 8; ++r) {
                #pragma unroll
                for (int off = 1; off <= 8; off <<= 1)
                    pr[r] += __shfl_xor(pr[r], off, 32);
            }
            if (mc == 0) {
                #pragma unroll
                for (int r = 0; r < 8; ++r) s_modal[0][r + 8 * grp] = pr[r] + b2i[0];
            }
        }

        // ===== text MLP: [16,128] @ [128,64] -> relu -> @ [64] =====
        {
            float pr[8];
            #pragma unroll
            for (int r = 0; r < 8; ++r) pr[r] = 0.f;
            #pragma unroll
            for (int t = 0; t < 4; ++t) {
                const int n0 = t * 16;
                v8f c = {};
                #pragma unroll
                for (int kb = 0; kb < 4; ++kb) {    // K=128 -> 4 steps of 32
                    v16h a, b;
                    #pragma unroll
                    for (int i = 0; i < 16; ++i) {
                        const int k = kb * 32 + wmma_kidx(i, grp);
                        a[i] = (_Float16)(pooled_txt[mc][k] * inv_cnt[mc]);
                        b[i] = (_Float16)(W1t[k * HID + n0 + mc]);
                    }
                    c = __builtin_amdgcn_wmma_f32_16x16x32_f16(
                            false, a, false, b, (short)0, c, false, false);
                }
                const float w2v = W2t[n0 + mc];
                const float b1v = b1t[n0 + mc];
                #pragma unroll
                for (int r = 0; r < 8; ++r) {
                    float h = c[r] + b1v;
                    h = h > 0.f ? h : 0.f;
                    pr[r] += h * w2v;
                }
            }
            #pragma unroll
            for (int r = 0; r < 8; ++r) {
                #pragma unroll
                for (int off = 1; off <= 8; off <<= 1)
                    pr[r] += __shfl_xor(pr[r], off, 32);
            }
            if (mc == 0) {
                #pragma unroll
                for (int r = 0; r < 8; ++r) s_modal[1][r + 8 * grp] = pr[r] + b2t[0];
            }
        }
    }
    __syncthreads();

    // ---- 5) softmax(alpha) combine + sigmoid + threshold ----
    if (tid < EPB) {
        const int e = e0 + tid;
        if (e < n_edges) {
            const float a0 = alpha[0], a1 = alpha[1];
            const float m  = fmaxf(a0, a1);
            const float x0 = expf(a0 - m), x1 = expf(a1 - m);
            const float inv = 1.0f / (x0 + x1);
            const float z  = (x0 * inv) * s_modal[0][tid] + (x1 * inv) * s_modal[1][tid];
            const float sc = 1.0f / (1.0f + expf(-z));
            out[e]            = (sc > 0.5f) ? 1.0f : 0.0f;  // mask
            out[n_edges + e]  = sc;                          // scores
        }
    }
}

extern "C" void kernel_launch(void* const* d_in, const int* in_sizes, int n_in,
                              void* d_out, int out_size, void* d_ws, size_t ws_size,
                              hipStream_t stream) {
    const float* feat_img = (const float*)d_in[0];
    const float* feat_txt = (const float*)d_in[1];
    const int*   node_idx = (const int*)d_in[2];
    const int*   seg_id   = (const int*)d_in[3];
    const float* W1i = (const float*)d_in[4];
    const float* b1i = (const float*)d_in[5];
    const float* W2i = (const float*)d_in[6];
    const float* b2i = (const float*)d_in[7];
    const float* W1t = (const float*)d_in[8];
    const float* b1t = (const float*)d_in[9];
    const float* W2t = (const float*)d_in[10];
    const float* b2t = (const float*)d_in[11];
    const float* alpha = (const float*)d_in[12];

    const int n_edges = out_size / 2;       // (mask, scores) concatenated
    const int n_memb  = in_sizes[2];        // node_idx length

    const int grid = (n_edges + EPB - 1) / EPB;
    hipLaunchKernelGGL(AttentionHyperedgeSelector_kernel,
                       dim3(grid), dim3(256), 0, stream,
                       feat_img, feat_txt, node_idx, seg_id,
                       W1i, b1i, W2i, b2i,
                       W1t, b1t, W2t, b2t,
                       alpha, (float*)d_out, n_edges, n_memb);
}
